// NPE_54202487275994
// MI455X (gfx1250) — matrix operand order repaired and checked by reading, not compiled, not run on hardware
//
#include <hip/hip_runtime.h>
#include <hip/hip_bf16.h>

typedef __attribute__((ext_vector_type(16))) _Float16 v16h;
typedef __attribute__((ext_vector_type(8)))  float    v8f;

#define NOBJ  32
#define BATCH 2048
#define NBT   128   // BATCH / 16
#define WPB   8     // waves per block in kernel 2

static __device__ inline v8f splat8(float x) {
    v8f v;
#pragma unroll
    for (int r = 0; r < 8; ++r) v[r] = x;
    return v;
}

static __device__ inline v16h splat16h(_Float16 x) {
    v16h v;
#pragma unroll
    for (int p = 0; p < 16; ++p) v[p] = x;
    return v;
}

// LeakyReLU(0.2) = max(x, 0.2*x)  -> v_pk_mul_f16 + v_pk_max_num_f16
static __device__ inline v16h lrelu16(v16h s) {
    return __builtin_elementwise_max(s, s * splat16h((_Float16)0.2f));
}

static __device__ inline v8f lrelu8(v8f s) {
    return __builtin_elementwise_max(s, s * splat8(0.2f));
}

// ---------------------------------------------------------------------------
// Kernel 1: per-row preprocessing.
//   A[row]  = x@W1[:20]              -> f16, WMMA A-operand layout
//   Bp[row] = x@W1[20:40] + b1       -> f16, WMMA A-operand layout
//   C0[row] = x@V1[:20]  + Vb1       -> f32, WMMA C/D-operand layout
//   mask[row] = x[16] > -1 ? 1 : 0
// A-operand layout (16x32 f16, wave32): lane = m + 16*((k>>3)&1),
// halfword pos = (k&7) | ((k>>4)<<3).  C/D layout: lane = (m>>3)*16 + (k&15),
// vgpr = (k>>4)*8 + (m&7).
// ---------------------------------------------------------------------------
__global__ __launch_bounds__(256) void npe_prep_kernel(
    const float* __restrict__ X,  const float* __restrict__ W1,
    const float* __restrict__ b1, const float* __restrict__ V1,
    const float* __restrict__ Vb1,
    _Float16* __restrict__ Abuf, _Float16* __restrict__ Bbuf,
    float* __restrict__ C0buf, float* __restrict__ maskbuf)
{
    __shared__ float w1s[40 * 32];
    __shared__ float v1s[20 * 32];
    __shared__ float b1s[32], vb1s[32];
    __shared__ float xs[8][20];

    const int tid = threadIdx.x;
    for (int idx = tid; idx < 40 * 32; idx += 256) w1s[idx] = W1[idx];
    for (int idx = tid; idx < 20 * 32; idx += 256) v1s[idx] = V1[idx];
    if (tid < 32) { b1s[tid] = b1[tid]; vb1s[tid] = Vb1[tid]; }

    const int tr  = tid >> 5;          // row within block (8 rows/block)
    const int k   = tid & 31;          // output channel
    const int row = blockIdx.x * 8 + tr;
    if (k < 20) xs[tr][k] = X[row * 20 + k];
    __syncthreads();

    float a = 0.f, bb = 0.f, cc = 0.f;
#pragma unroll
    for (int f = 0; f < 20; ++f) {
        const float xv = xs[tr][f];
        a  += xv * w1s[f * 32 + k];
        bb += xv * w1s[(20 + f) * 32 + k];
        cc += xv * v1s[f * 32 + k];
    }
    bb += b1s[k];
    cc += vb1s[k];

    const int obj = row >> 11;       // / BATCH
    const int b   = row & 2047;
    const int bt  = b >> 4;
    const int m   = b & 15;

    // f16 A-operand layout scatter
    const int laneA = m + 16 * ((k >> 3) & 1);
    const int posA  = (k & 7) | ((k >> 4) << 3);
    const int idxA  = ((obj * NBT + bt) * 32 + laneA) * 16 + posA;
    Abuf[idxA] = (_Float16)a;
    Bbuf[idxA] = (_Float16)bb;

    // f32 C/D-operand layout scatter
    const int laneC = (m >> 3) * 16 + (k & 15);
    const int posC  = (k >> 4) * 8 + (m & 7);
    C0buf[((obj * NBT + bt) * 32 + laneC) * 16 + posC] = cc;

    if (k == 16) maskbuf[row] = (xs[tr][16] > -1.f) ? 1.f : 0.f;
}

// ---------------------------------------------------------------------------
// Kernel 2: one wave per (i, batch-tile-of-16).
//   S = sum_{j!=i} mask_j * lrelu( lrelu(A_i + B_j) @ W2 + b2 )    [31x 2 WMMA]
//   val = S@W3 + cnt*b3 ; g = lrelu(C0 + val@V1bot) ; g = lrelu(g@V2+Vb2)
//   y = tanh(g@V3pad + Vb3) ; out = mask_i*y - (1-mask_i)*2        [+7 WMMA]
// ---------------------------------------------------------------------------
__global__ __launch_bounds__(256) void npe_main_kernel(
    const _Float16* __restrict__ Abuf, const _Float16* __restrict__ Bbuf,
    const float* __restrict__ C0buf,   const float* __restrict__ maskbuf,
    const float* __restrict__ W2, const float* __restrict__ b2,
    const float* __restrict__ W3, const float* __restrict__ b3,
    const float* __restrict__ V1, const float* __restrict__ Vb2,
    const float* __restrict__ V2, const float* __restrict__ V3,
    const float* __restrict__ Vb3,
    float* __restrict__ out)
{
    __shared__ float smem[WPB][1024];   // [0..511]=mask tile, [512..1023]=transpose ws

    const int tid  = threadIdx.x;
    const int wave = tid >> 5;
    const int lane = tid & 31;
    const int tile = blockIdx.x * WPB + wave;
    const int i  = tile >> 7;           // / NBT
    const int bt = tile & (NBT - 1);
    const int c  = lane & 15;           // column within 16-wide D tile
    const int hi = lane >> 4;           // row-half selector

    float* ml  = &smem[wave][0];
    float* wsx = &smem[wave][512];

    // ---- gather weights into WMMA B-operand layout (f16) ----
    // B-layout (32x16 f16): pos p<8 -> K=8*hi+p, pos p>=8 -> K=16+8*hi+(p-8);
    // column = lane&15.
    v16h w2t0, w2t1, w3t0, w3t1, v1b0, v1b1, v2t0, v2t1, v3t;
#pragma unroll
    for (int p = 0; p < 16; ++p) {
        const int K = (p < 8) ? (hi * 8 + p) : (16 + hi * 8 + (p - 8));
        w2t0[p] = (_Float16)W2[K * 32 + c];
        w2t1[p] = (_Float16)W2[K * 32 + 16 + c];
        w3t0[p] = (_Float16)W3[K * 32 + c];
        w3t1[p] = (_Float16)W3[K * 32 + 16 + c];
        v1b0[p] = (_Float16)V1[(20 + K) * 32 + c];
        v1b1[p] = (_Float16)V1[(20 + K) * 32 + 16 + c];
        v2t0[p] = (_Float16)V2[K * 32 + c];
        v2t1[p] = (_Float16)V2[K * 32 + 16 + c];
        v3t[p]  = (c < 2) ? (_Float16)V3[K * 2 + c] : (_Float16)0.f;
    }
    const float b2c0 = b2[c],  b2c1 = b2[16 + c];
    const float b3c0 = b3[c],  b3c1 = b3[16 + c];
    const float vb2c0 = Vb2[c], vb2c1 = Vb2[16 + c];
    const float vb3c  = (c < 2) ? Vb3[c] : 0.f;

    // ---- stage mask tile for this (bt): ml[j*16 + m] ----
#pragma unroll
    for (int t = 0; t < 16; ++t) {
        const int idx = lane + 32 * t;
        const int j = idx >> 4, m = idx & 15;
        ml[idx] = maskbuf[j * BATCH + bt * 16 + m];
    }

    // ---- A_i tile (already in A-operand layout, f16) ----
    const v16h* Ab = (const v16h*)Abuf;
    const v16h* Bb = (const v16h*)Bbuf;
    const v16h  ai = Ab[(i * NBT + bt) * 32 + lane];

    v8f S0 = splat8(0.f), S1 = splat8(0.f);
    float cnt[8];
#pragma unroll
    for (int r = 0; r < 8; ++r) cnt[r] = 0.f;
    const v8f cb20 = splat8(b2c0), cb21 = splat8(b2c1);

    for (int j = 0; j < NOBJ; ++j) {
        if (j == i) continue;
        const v16h bj = Bb[(j * NBT + bt) * 32 + lane];
        const v16h h  = lrelu16(ai + bj);                 // layer-1 (factored)
        v8f d0 = __builtin_amdgcn_wmma_f32_16x16x32_f16(
            false, h, false, w2t0, (short)0, cb20, false, false);
        v8f d1 = __builtin_amdgcn_wmma_f32_16x16x32_f16(
            false, h, false, w2t1, (short)0, cb21, false, false);
        d0 = lrelu8(d0);
        d1 = lrelu8(d1);
        const float* mp = &ml[j * 16 + hi * 8];
#pragma unroll
        for (int r = 0; r < 8; ++r) {
            const float mr = mp[r];
            S0[r] += mr * d0[r];
            S1[r] += mr * d1[r];
            cnt[r] += mr;
        }
    }

    // ---- D-layout f32 -> A-operand f16 via per-wave LDS scratch ----
    auto d2a = [&](const v8f& D0, const v8f& D1) -> v16h {
#pragma unroll
        for (int r = 0; r < 8; ++r) {
            wsx[(8 * hi + r) * 32 + c]      = D0[r];
            wsx[(8 * hi + r) * 32 + 16 + c] = D1[r];
        }
        v16h A;
        const int base = c * 32 + hi * 8;   // row m=c, K offset hi*8
#pragma unroll
        for (int p = 0; p < 8; ++p) {
            A[p]     = (_Float16)wsx[base + p];
            A[8 + p] = (_Float16)wsx[base + 16 + p];
        }
        return A;
    };

    // val = S@W3 + cnt*b3
    const v16h sA = d2a(S0, S1);
    v8f cb30, cb31;
#pragma unroll
    for (int r = 0; r < 8; ++r) { cb30[r] = cnt[r] * b3c0; cb31[r] = cnt[r] * b3c1; }
    v8f val0 = __builtin_amdgcn_wmma_f32_16x16x32_f16(
        false, sA, false, w3t0, (short)0, cb30, false, false);
    v8f val1 = __builtin_amdgcn_wmma_f32_16x16x32_f16(
        false, sA, false, w3t1, (short)0, cb31, false, false);

    // g1 = lrelu( C0 + val@V1[20:52] )   (C0 = x@V1[:20]+Vb1, in C/D layout)
    const v16h vA = d2a(val0, val1);
    const float* C0p = &C0buf[((i * NBT + bt) * 32 + lane) * 16];
    v8f c00, c01;
#pragma unroll
    for (int r = 0; r < 8; ++r) { c00[r] = C0p[r]; c01[r] = C0p[8 + r]; }
    v8f g0 = __builtin_amdgcn_wmma_f32_16x16x32_f16(
        false, vA, false, v1b0, (short)0, c00, false, false);
    v8f g1 = __builtin_amdgcn_wmma_f32_16x16x32_f16(
        false, vA, false, v1b1, (short)0, c01, false, false);
    g0 = lrelu8(g0);
    g1 = lrelu8(g1);

    // g2 = lrelu(g1@V2 + Vb2)
    const v16h gA = d2a(g0, g1);
    v8f u0 = __builtin_amdgcn_wmma_f32_16x16x32_f16(
        false, gA, false, v2t0, (short)0, splat8(vb2c0), false, false);
    v8f u1 = __builtin_amdgcn_wmma_f32_16x16x32_f16(
        false, gA, false, v2t1, (short)0, splat8(vb2c1), false, false);
    u0 = lrelu8(u0);
    u1 = lrelu8(u1);

    // y = tanh(g2@V3pad + Vb3) ; only columns 0,1 of the padded tile are real
    const v16h g2A = d2a(u0, u1);
    const v8f y = __builtin_amdgcn_wmma_f32_16x16x32_f16(
        false, g2A, false, v3t, (short)0, splat8(vb3c), false, false);

    if (c < 2) {
#pragma unroll
        for (int r = 0; r < 8; ++r) {
            const int m = 8 * hi + r;
            const float mk = ml[i * 16 + m];
            const float g  = tanhf(y[r]);
            out[(i * BATCH + bt * 16 + m) * 2 + c] = mk * g - (1.f - mk) * 2.f;
        }
    }
}

extern "C" void kernel_launch(void* const* d_in, const int* in_sizes, int n_in,
                              void* d_out, int out_size, void* d_ws, size_t ws_size,
                              hipStream_t stream) {
    (void)in_sizes; (void)n_in; (void)out_size; (void)ws_size;
    const float* X   = (const float*)d_in[0];
    const float* W1  = (const float*)d_in[1];
    const float* b1  = (const float*)d_in[2];
    const float* W2  = (const float*)d_in[3];
    const float* b2  = (const float*)d_in[4];
    const float* W3  = (const float*)d_in[5];
    const float* b3  = (const float*)d_in[6];
    const float* V1  = (const float*)d_in[7];
    const float* Vb1 = (const float*)d_in[8];
    const float* V2  = (const float*)d_in[9];
    const float* Vb2 = (const float*)d_in[10];
    const float* V3  = (const float*)d_in[11];
    const float* Vb3 = (const float*)d_in[12];
    float* out = (float*)d_out;

    // workspace layout
    char* ws = (char*)d_ws;
    _Float16* Abuf   = (_Float16*)(ws);                      // 4 MB
    _Float16* Bbuf   = (_Float16*)(ws + (4u << 20));         // 4 MB
    float*    C0buf  = (float*)   (ws + (8u << 20));         // 8 MB
    float*    maskbf = (float*)   (ws + (16u << 20));        // 256 KB

    const int rows = NOBJ * BATCH;                           // 65536
    npe_prep_kernel<<<rows / 8, 256, 0, stream>>>(
        X, W1, b1, V1, Vb1, Abuf, Bbuf, C0buf, maskbf);

    const int tiles = NOBJ * NBT;                            // 4096 waves
    npe_main_kernel<<<tiles / WPB, 256, 0, stream>>>(
        Abuf, Bbuf, C0buf, maskbf,
        W2, b2, W3, b3, V1, Vb2, V2, V3, Vb3, out);
}
